// SlotAttention_1692217115510
// MI455X (gfx1250) — matrix operand order repaired and checked by reading, not compiled, not run on hardware
//
#include <hip/hip_runtime.h>

// ---------------------------------------------------------------------------
// SlotAttention for MI455X (gfx1250).
//  Phase 0: convert all weight matrices f32 -> bf16 into workspace.
//  Phase 1: parallel LN(x) + K/V projection for all B*NF*N rows using
//           v_wmma_f32_16x16x32_bf16; results staged in LDS and written with
//           coalesced b128 stores. K/V stored bf16 in workspace (128 MB,
//           L2-resident at 192 MB).
//  Phase 2: persistent scan kernel, grid = B (independent recurrent chains),
//           16 waves / block; q-proj, dots, softmax-over-slots, renorm,
//           updates, GRU, FFN all via WMMA bf16 with f32 accumulation.
//           Next frame's K/V prefetched (global_prefetch_b8) during dots.
// Workspace layout (elements of u16): K[B*NF*N*D], V[B*NF*N*D],
//   Wq[D*D], Wk[D*D], Wv[D*D], Wih[3D*D], Whh[3D*D], W1[H*D], W2[D*H]
// Total ~128.4 MB.
// ---------------------------------------------------------------------------

#define DEV __device__ __forceinline__

static constexpr int Bb = 16, NF = 16, Nn = 1024, Dd = 128, Ss = 8, Hh = 128;
static constexpr float SCALE = 0.08838834764831845f;   // 128^-0.5

typedef __attribute__((ext_vector_type(16))) __bf16 v16bf;
typedef __attribute__((ext_vector_type(8)))  float  v8f;

union FragU { v16bf bf; unsigned int u[8]; };

DEV unsigned short f2bf(float f) {            // round-to-nearest-even f32->bf16
  unsigned int x = __float_as_uint(f);
  x += 0x7fffu + ((x >> 16) & 1u);
  return (unsigned short)(x >> 16);
}

// A fragment (16x32 bf16, MxK): lane m=lane&15; VGPR v holds K pair at
// khalf+2v (khalf = 0 / 16 for lane<16 / >=16).
DEV v16bf load_a_frag(const unsigned short* p, int stride) {
  const int lane  = threadIdx.x & 31;
  const int m     = lane & 15;
  const int khalf = (lane >> 4) << 4;
  const unsigned short* row = p + m * stride + khalf;
  FragU f;
#pragma unroll
  for (int v = 0; v < 8; ++v) f.u[v]     = *(const unsigned int*)(row + 2 * v);
  return f.bf;
}

// B fragment (32x16 bf16, KxN) from memory laid out [n][k] (k contiguous):
// lane n=lane&15; lanes<16 cover K 0..15, lanes>=16 cover K 16..31.
DEV v16bf load_b_frag_nk(const unsigned short* p, int stride) {
  const int lane = threadIdx.x & 31;
  const int n    = lane & 15;
  const int kh   = (lane >> 4) << 4;
  const unsigned short* q = p + n * stride + kh;
  FragU f;
#pragma unroll
  for (int v = 0; v < 8; ++v) f.u[v] = *(const unsigned int*)(q + 2 * v);
  return f.bf;
}

// B fragment from memory laid out [k][n] (n contiguous) -> two u16 gathers.
DEV v16bf load_b_frag_kn(const unsigned short* p, int stride) {
  const int lane = threadIdx.x & 31;
  const int n    = lane & 15;
  const int kh   = (lane >> 4) << 4;
  FragU f;
#pragma unroll
  for (int v = 0; v < 8; ++v) {
    unsigned int lo = p[(kh + 2 * v    ) * stride + n];
    unsigned int hi = p[(kh + 2 * v + 1) * stride + n];
    f.u[v] = lo | (hi << 16);
  }
  return f.bf;
}

// 16x16 tile GEMM over K (multiple of 32): A from LDS, B row-major [n][k].
DEV v8f wmma_k(const unsigned short* A, int as, const unsigned short* B, int bs,
               int K, v8f acc) {
#pragma unroll 4
  for (int k0 = 0; k0 < K; k0 += 32) {
    v16bf a = load_a_frag(A + k0, as);
    v16bf b = load_b_frag_nk(B + k0, bs);
    acc = __builtin_amdgcn_wmma_f32_16x16x32_bf16(false, a, false, b,
                                                  (short)0, acc, false, false);
  }
  return acc;
}

// ---------------------------------------------------------------------------
__global__ __launch_bounds__(256) void k_cvt(const float* __restrict__ s,
                                             unsigned short* __restrict__ d, int n) {
  int i = blockIdx.x * 256 + threadIdx.x;
  if (i < n) d[i] = f2bf(s[i]);
}

// ---------------------------------------------------------------------------
// Phase 1: 64 rows/block. LN -> LDS bf16 (stride 136 avoids bank conflicts),
// then K and V projections (32 tiles each over 8 waves); D-fragments staged in
// LDS and written back with coalesced 16-byte stores.
__global__ __launch_bounds__(256) void k_ln_kv(
    const float* __restrict__ x, const float* __restrict__ g,
    const float* __restrict__ be,
    const unsigned short* __restrict__ wk, const float* __restrict__ bk,
    const unsigned short* __restrict__ wv, const float* __restrict__ bv,
    unsigned short* __restrict__ K, unsigned short* __restrict__ V) {
  __shared__ unsigned short xn[64][136];
  __shared__ unsigned short stg[64][136];
  const int wave = threadIdx.x >> 5, lane = threadIdx.x & 31;
  const long rowbase = (long)blockIdx.x * 64;

  const float4 gv  = ((const float4*)g)[lane];
  const float4 bev = ((const float4*)be)[lane];
  for (int r = wave; r < 64; r += 8) {
    const float4 xv = ((const float4*)(x + (rowbase + r) * Dd))[lane];
    float s  = xv.x + xv.y + xv.z + xv.w;
    float s2 = xv.x * xv.x + xv.y * xv.y + xv.z * xv.z + xv.w * xv.w;
#pragma unroll
    for (int o = 16; o; o >>= 1) { s += __shfl_xor(s, o, 32); s2 += __shfl_xor(s2, o, 32); }
    float m    = s * (1.f / 128.f);
    float rstd = rsqrtf(s2 * (1.f / 128.f) - m * m + 1e-5f);
    int c = lane * 4;
    xn[r][c + 0] = f2bf((xv.x - m) * rstd * gv.x + bev.x);
    xn[r][c + 1] = f2bf((xv.y - m) * rstd * gv.y + bev.y);
    xn[r][c + 2] = f2bf((xv.z - m) * rstd * gv.z + bev.z);
    xn[r][c + 3] = f2bf((xv.w - m) * rstd * gv.w + bev.w);
  }
  __syncthreads();

  for (int mat = 0; mat < 2; ++mat) {              // 0 = K, 1 = V
    const unsigned short* W    = mat ? wv : wk;
    const float*          bias = mat ? bv : bk;
    unsigned short*       outp = mat ? V : K;
#pragma unroll
    for (int t = 0; t < 4; ++t) {
      int tile = wave * 4 + t;                     // 0..31
      int mt   = tile >> 3;
      int nt   = tile & 7;
      v8f acc = {};
      acc = wmma_k(&xn[mt * 16][0], 136, W + nt * 16 * Dd, Dd, Dd, acc);
      int   col = nt * 16 + (lane & 15);
      float bj  = bias[col];
      int   r0  = mt * 16 + ((lane >> 4) << 3);
#pragma unroll
      for (int r = 0; r < 8; ++r) stg[r0 + r][col] = f2bf(acc[r] + bj);
    }
    __syncthreads();
    // coalesced writeback: 64x128 u16, 16B per thread, contiguous
    for (int i = threadIdx.x; i < 64 * 128 / 8; i += 256) {
      int row = i >> 4, c8 = (i & 15) * 8;
      uint4 dv = *(const uint4*)&stg[row][c8];
      *(uint4*)&outp[(rowbase + row) * Dd + c8] = dv;
    }
    __syncthreads();
  }
}

// ---------------------------------------------------------------------------
// Phase 2: one block per batch element, 512 threads (16 waves), 17 iterations.
__global__ __launch_bounds__(512) void k_scan(
    const unsigned short* __restrict__ Kbf, const unsigned short* __restrict__ Vbf,
    const unsigned short* __restrict__ wq,  const unsigned short* __restrict__ wih,
    const unsigned short* __restrict__ whh, const unsigned short* __restrict__ w1,
    const unsigned short* __restrict__ w2,
    const float* __restrict__ bq, const float* __restrict__ bih,
    const float* __restrict__ bhh, const float* __restrict__ b1,
    const float* __restrict__ b2, const float* __restrict__ slots_init,
    const float* __restrict__ g_sl, const float* __restrict__ be_sl,
    const float* __restrict__ g_ff, const float* __restrict__ be_ff,
    float* __restrict__ out) {
  __shared__ __align__(16) float slots[8][128];
  __shared__ unsigned short sA[16][136];     // A operand buffer (rows 8..15 = 0)
  __shared__ unsigned short sB[16][136];     // A operand buffer (rows 8..15 = 0)
  __shared__ float rowsum[8];
  union ShmU {
    unsigned short attn[16][1032];                       // 33 KB
    struct { float gi[8][384]; float gh[8][384]; } g;    // 24 KB (overlay)
  };
  __shared__ ShmU u;

  const int tid = threadIdx.x, wave = tid >> 5, lane = tid & 31;
  const int b = blockIdx.x;
  float* slotsOut = out;
  float* attnOut  = out + (size_t)Bb * NF * Ss * Dd;

  for (int i = tid; i < 16 * 136; i += 512) { (&sA[0][0])[i] = 0; (&sB[0][0])[i] = 0; }
  for (int i = tid; i < Ss * Dd; i += 512)  slots[i >> 7][i & 127] = slots_init[i];
  __syncthreads();

  for (int t = 0; t < NF + 1; ++t) {               // t==0: warm-up on frame 0
    const int  f      = (t == 0) ? 0 : (t - 1);
    const bool record = (t > 0);
    const unsigned short* Kf = Kbf + (size_t)(b * NF + f) * Nn * Dd;
    const unsigned short* Vf = Vbf + (size_t)(b * NF + f) * Nn * Dd;
    const int  fn     = (t < NF) ? t : NF - 1;     // next iteration's frame
    const unsigned short* Kn = Kbf + (size_t)(b * NF + fn) * Nn * Dd;
    const unsigned short* Vn = Vbf + (size_t)(b * NF + fn) * Nn * Dd;

    if (tid < 8) rowsum[tid] = 0.f;
    // ---- LN(slots) with g_sl -> sA -------------------------------------
    if (wave < 8) {
      int s = wave;
      float4 xv = ((const float4*)&slots[s][0])[lane];
      float sm = xv.x + xv.y + xv.z + xv.w;
      float s2 = xv.x * xv.x + xv.y * xv.y + xv.z * xv.z + xv.w * xv.w;
#pragma unroll
      for (int o = 16; o; o >>= 1) { sm += __shfl_xor(sm, o, 32); s2 += __shfl_xor(s2, o, 32); }
      float m = sm * (1.f / 128.f), rstd = rsqrtf(s2 * (1.f / 128.f) - m * m + 1e-5f);
      float4 gv = ((const float4*)g_sl)[lane], bev = ((const float4*)be_sl)[lane];
      int c = lane * 4;
      sA[s][c + 0] = f2bf((xv.x - m) * rstd * gv.x + bev.x);
      sA[s][c + 1] = f2bf((xv.y - m) * rstd * gv.y + bev.y);
      sA[s][c + 2] = f2bf((xv.z - m) * rstd * gv.z + bev.z);
      sA[s][c + 3] = f2bf((xv.w - m) * rstd * gv.w + bev.w);
    }
    __syncthreads();
    // ---- q = LN(slots) @ Wq^T + bq -> sB (bf16) ------------------------
    if (wave < 8) {
      v8f acc = {};
      acc = wmma_k(&sA[0][0], 136, wq + wave * 16 * Dd, Dd, Dd, acc);
      if (lane < 16) {
        int col = wave * 16 + lane;
        float bj = bq[col];
#pragma unroll
        for (int r = 0; r < 8; ++r) sB[r][col] = f2bf(acc[r] + bj);
      }
    }
    __syncthreads();
    // ---- dots = q @ K^T, softmax over slots (in-lane), attn_ori out ----
    float aor[4][8];
    float psum[8];
#pragma unroll
    for (int r = 0; r < 8; ++r) psum[r] = 0.f;
#pragma unroll
    for (int i = 0; i < 4; ++i) {
      int nt = wave * 4 + i;                       // 64 n-tiles over 16 waves
      // prefetch next frame's K/V rows for this tile range (global_prefetch_b8)
      int pr = (nt * 16 + lane) & (Nn - 1);
      __builtin_prefetch(Kn + (size_t)pr * Dd, 0, 1);
      __builtin_prefetch(Vn + (size_t)pr * Dd, 0, 1);
      v8f acc = {};
      acc = wmma_k(&sB[0][0], 136, Kf + nt * 16 * Dd, Dd, Dd, acc);
      if (lane < 16) {                             // lanes 0-15 hold s = 0..7
        int n = nt * 16 + lane;
        float dv[8], mx = -1e30f;
#pragma unroll
        for (int r = 0; r < 8; ++r) { dv[r] = acc[r] * SCALE; mx = fmaxf(mx, dv[r]); }
        float es = 0.f;
#pragma unroll
        for (int r = 0; r < 8; ++r) { dv[r] = __expf(dv[r] - mx); es += dv[r]; }
        float inv = 1.f / es;
#pragma unroll
        for (int r = 0; r < 8; ++r) {
          float ao = dv[r] * inv + 1e-8f;          // attn_ori (softmax + EPS)
          aor[i][r] = ao; psum[r] += ao;
          if (record) attnOut[((size_t)(b * NF + f) * Ss + r) * Nn + n] = ao;
        }
      }
    }
#pragma unroll
    for (int r = 0; r < 8; ++r) {
#pragma unroll
      for (int o = 1; o < 16; o <<= 1) psum[r] += __shfl_xor(psum[r], o, 32);
    }
    if (lane == 0) {
#pragma unroll
      for (int r = 0; r < 8; ++r) atomicAdd(&rowsum[r], psum[r]);
    }
    __syncthreads();
    // ---- renormalize over N, store attn bf16 ---------------------------
    {
      float inv[8];
#pragma unroll
      for (int r = 0; r < 8; ++r) inv[r] = 1.f / rowsum[r];
      if (lane < 16) {
#pragma unroll
        for (int i = 0; i < 4; ++i) {
          int n = (wave * 4 + i) * 16 + lane;
#pragma unroll
          for (int r = 0; r < 8; ++r) u.attn[r][n] = f2bf(aor[i][r] * inv[r]);
        }
      }
    }
    __syncthreads();
    // ---- updates = attn @ V (waves 0-7); slots -> bf16 (waves 8-15) ----
    if (wave < 8) {
      v8f acc = {};
      const int col0 = wave * 16;
      for (int k0 = 0; k0 < Nn; k0 += 32) {
        v16bf a  = load_a_frag(&u.attn[0][k0], 1032);
        v16bf bb = load_b_frag_kn(Vf + (size_t)k0 * Dd + col0, Dd);
        acc = __builtin_amdgcn_wmma_f32_16x16x32_bf16(false, a, false, bb,
                                                      (short)0, acc, false, false);
      }
      if (lane < 16) {
        int col = col0 + lane;
#pragma unroll
        for (int r = 0; r < 8; ++r) sA[r][col] = f2bf(acc[r]);
      }
    } else {
      for (int i = tid - 256; i < Ss * Dd; i += 256)
        sB[i >> 7][i & 127] = f2bf(slots[i >> 7][i & 127]);
    }
    __syncthreads();
    // ---- GRU gates: gi = upd@Wih^T+bih, gh = slots@Whh^T+bhh -----------
#pragma unroll
    for (int i = 0; i < 3; ++i) {
      int tile = wave * 3 + i;                     // 48 tiles over 16 waves
      int mat  = tile & 1;                         // 0: gi, 1: gh
      int nt   = tile >> 1;                        // 0..23
      const unsigned short* A = mat ? &sB[0][0] : &sA[0][0];
      const unsigned short* W = mat ? whh : wih;
      const float*       bias = mat ? bhh : bih;
      v8f acc = {};
      acc = wmma_k(A, 136, W + nt * 16 * Dd, Dd, Dd, acc);
      if (lane < 16) {
        int col = nt * 16 + lane;
        float* dst = mat ? &u.g.gh[0][0] : &u.g.gi[0][0];
        float bj = bias[col];
#pragma unroll
        for (int r = 0; r < 8; ++r) dst[r * 384 + col] = acc[r] + bj;
      }
    }
    __syncthreads();
    // ---- GRU elementwise (in place) ------------------------------------
    for (int e = tid; e < Ss * Dd; e += 512) {
      int s = e >> 7, j = e & 127;
      float r  = 1.f / (1.f + __expf(-(u.g.gi[s][j]       + u.g.gh[s][j])));
      float z  = 1.f / (1.f + __expf(-(u.g.gi[s][128 + j] + u.g.gh[s][128 + j])));
      float nn = tanhf(u.g.gi[s][256 + j] + r * u.g.gh[s][256 + j]);
      slots[s][j] = (1.f - z) * nn + z * slots[s][j];
    }
    __syncthreads();
    // ---- FFN: LN with g_ff -> sA ---------------------------------------
    if (wave < 8) {
      int s = wave;
      float4 xv = ((const float4*)&slots[s][0])[lane];
      float sm = xv.x + xv.y + xv.z + xv.w;
      float s2 = xv.x * xv.x + xv.y * xv.y + xv.z * xv.z + xv.w * xv.w;
#pragma unroll
      for (int o = 16; o; o >>= 1) { sm += __shfl_xor(sm, o, 32); s2 += __shfl_xor(s2, o, 32); }
      float m = sm * (1.f / 128.f), rstd = rsqrtf(s2 * (1.f / 128.f) - m * m + 1e-5f);
      float4 gv = ((const float4*)g_ff)[lane], bev = ((const float4*)be_ff)[lane];
      int c = lane * 4;
      sA[s][c + 0] = f2bf((xv.x - m) * rstd * gv.x + bev.x);
      sA[s][c + 1] = f2bf((xv.y - m) * rstd * gv.y + bev.y);
      sA[s][c + 2] = f2bf((xv.z - m) * rstd * gv.z + bev.z);
      sA[s][c + 3] = f2bf((xv.w - m) * rstd * gv.w + bev.w);
    }
    __syncthreads();
    // ---- h1 = relu(sA @ W1^T + b1) -> sB -------------------------------
    if (wave < 8) {
      v8f acc = {};
      acc = wmma_k(&sA[0][0], 136, w1 + wave * 16 * Dd, Dd, Dd, acc);
      if (lane < 16) {
        int col = wave * 16 + lane;
        float bj = b1[col];
#pragma unroll
        for (int r = 0; r < 8; ++r) sB[r][col] = f2bf(fmaxf(acc[r] + bj, 0.f));
      }
    }
    __syncthreads();
    // ---- ff = h1 @ W2^T + b2, residual, store --------------------------
    if (wave < 8) {
      v8f acc = {};
      acc = wmma_k(&sB[0][0], 136, w2 + wave * 16 * Hh, Hh, Hh, acc);
      if (lane < 16) {
        int col = wave * 16 + lane;
        float bj = b2[col];
#pragma unroll
        for (int r = 0; r < 8; ++r) {
          float res = slots[r][col] + acc[r] + bj;
          slots[r][col] = res;
          if (record) slotsOut[((size_t)(b * NF + f) * Ss + r) * Dd + col] = res;
        }
      }
    }
    __syncthreads();
  }
}

// ---------------------------------------------------------------------------
extern "C" void kernel_launch(void* const* d_in, const int* in_sizes, int n_in,
                              void* d_out, int out_size, void* d_ws, size_t ws_size,
                              hipStream_t stream) {
  (void)in_sizes; (void)n_in; (void)out_size; (void)ws_size;
  const float* inputs     = (const float*)d_in[0];
  const float* slots_init = (const float*)d_in[1];
  const float* Wq  = (const float*)d_in[2];  const float* bq  = (const float*)d_in[3];
  const float* Wk  = (const float*)d_in[4];  const float* bk  = (const float*)d_in[5];
  const float* Wv  = (const float*)d_in[6];  const float* bv  = (const float*)d_in[7];
  const float* W1  = (const float*)d_in[8];  const float* b1  = (const float*)d_in[9];
  const float* W2  = (const float*)d_in[10]; const float* b2  = (const float*)d_in[11];
  const float* Wih = (const float*)d_in[12]; const float* Whh = (const float*)d_in[13];
  const float* bih = (const float*)d_in[14]; const float* bhh = (const float*)d_in[15];
  const float* g_in  = (const float*)d_in[16]; const float* be_in = (const float*)d_in[17];
  const float* g_sl  = (const float*)d_in[18]; const float* be_sl = (const float*)d_in[19];
  const float* g_ff  = (const float*)d_in[20]; const float* be_ff = (const float*)d_in[21];

  const size_t kvElems = (size_t)Bb * NF * Nn * Dd;      // 33,554,432
  unsigned short* Kbf = (unsigned short*)d_ws;
  unsigned short* Vbf = Kbf + kvElems;
  unsigned short* wq  = Vbf + kvElems;
  unsigned short* wk  = wq  + Dd * Dd;
  unsigned short* wv  = wk  + Dd * Dd;
  unsigned short* wih = wv  + Dd * Dd;
  unsigned short* whh = wih + 3 * Dd * Dd;
  unsigned short* w1  = whh + 3 * Dd * Dd;
  unsigned short* w2  = w1  + Hh * Dd;

  k_cvt<<<64,  256, 0, stream>>>(Wq,  wq,  Dd * Dd);
  k_cvt<<<64,  256, 0, stream>>>(Wk,  wk,  Dd * Dd);
  k_cvt<<<64,  256, 0, stream>>>(Wv,  wv,  Dd * Dd);
  k_cvt<<<192, 256, 0, stream>>>(Wih, wih, 3 * Dd * Dd);
  k_cvt<<<192, 256, 0, stream>>>(Whh, whh, 3 * Dd * Dd);
  k_cvt<<<64,  256, 0, stream>>>(W1,  w1,  Hh * Dd);
  k_cvt<<<64,  256, 0, stream>>>(W2,  w2,  Dd * Hh);

  k_ln_kv<<<(int)(kvElems / Dd / 64), 256, 0, stream>>>(
      inputs, g_in, be_in, wk, bk, wv, bv, Kbf, Vbf);

  k_scan<<<Bb, 512, 0, stream>>>(Kbf, Vbf, wq, wih, whh, w1, w2,
                                 bq, bih, bhh, b1, b2, slots_init,
                                 g_sl, be_sl, g_ff, be_ff, (float*)d_out);
}
